// SpatialAxialAttention_58471684768287
// MI455X (gfx1250) — matrix operand order repaired and checked by reading, not compiled, not run on hardware
//
#include <hip/hip_runtime.h>
#include <hip/hip_bf16.h>

typedef __bf16 bf16;
typedef __attribute__((ext_vector_type(8)))  bf16  v8bf;
typedef __attribute__((ext_vector_type(16))) bf16  v16bf;
typedef __attribute__((ext_vector_type(8)))  float v8f;

#define Bc 8
#define Cc 256
#define Hc 128
#define Wc 128
#define Rc 32
#define LDW 264   // padded leading dim (halves) for 256-wide tiles: 528B = 33*16
#define LDQ 40    // padded ld for 32-wide q/k tiles: 80B = 5*16
#define LDV 136   // padded ld for 128-wide tiles: 272B = 17*16
#define LDS_SC 132 // f32 score row stride: 528B

// ---------------------------------------------------------------------------
// WMMA helpers (CDNA5 wave32, v_wmma_f32_16x16x32_bf16)
// ---------------------------------------------------------------------------
__device__ __forceinline__ v8f wmma_bf16(v16bf a, v16bf b, v8f c) {
  // 8 args: (neg_a, A, neg_b, B, c_mod, C, reuse_a, reuse_b)
  return __builtin_amdgcn_wmma_f32_16x16x32_bf16(false, a, false, b, (short)0, c,
                                                 false, false);
}

// A fragment: 16x32 bf16 tile stored row-major [m][k] in LDS.
// ISA layout: lane L (m = L&15, hi = L>>4) holds K in {hi*8..hi*8+7} U {16+hi*8..16+hi*8+7}.
__device__ __forceinline__ v16bf frag_a(const bf16* __restrict__ tile, int ld, int k0,
                                        int lane) {
  const int m = lane & 15, hi = lane >> 4;
  const bf16* p = tile + m * ld + k0 + hi * 8;
  v8bf lo = *(const v8bf*)p;
  v8bf hi8 = *(const v8bf*)(p + 16);
  return __builtin_shufflevector(lo, hi8, 0, 1, 2, 3, 4, 5, 6, 7, 8, 9, 10, 11, 12, 13,
                                 14, 15);
}

// B fragment: 32x16 bf16. We store B transposed (N-major [n][k]) so each lane reads
// 2x8 contiguous halves. ISA layout: lane L (n = L&15, hi = L>>4) holds K in {hi*8..hi*8+7} U {16+hi*8..16+hi*8+7}.
__device__ __forceinline__ v16bf frag_b(const bf16* __restrict__ tileT, int ld, int k0,
                                        int lane) {
  const int n = lane & 15, hi = lane >> 4;
  const bf16* p = tileT + n * ld + k0 + hi * 8;
  v8bf a = *(const v8bf*)p;
  v8bf b = *(const v8bf*)(p + 16);
  return __builtin_shufflevector(a, b, 0, 1, 2, 3, 4, 5, 6, 7, 8, 9, 10, 11, 12, 13, 14,
                                 15);
}

// ---------------------------------------------------------------------------
// Kernel 1: QKV projection. One workgroup (256 threads = 8 waves) per (b,h) line.
// ---------------------------------------------------------------------------
__global__ void qkv_kernel(const float* __restrict__ x, const float* __restrict__ Wq,
                           const float* __restrict__ bq, const float* __restrict__ Wk,
                           const float* __restrict__ bk, const float* __restrict__ Wv,
                           const float* __restrict__ bv, bf16* __restrict__ qws,
                           bf16* __restrict__ kws, bf16* __restrict__ vws) {
  extern __shared__ char smem[];
  bf16* xlds = (bf16*)smem;                 // [128][LDW] : X^T (w-major)
  bf16* wq = xlds + Wc * LDW;               // [32][LDW]
  bf16* wk = wq + Rc * LDW;                 // [32][LDW]
  bf16* wv = wk + Rc * LDW;                 // [256][LDW]

  const int bid = blockIdx.x;
  const int b = bid >> 7, h = bid & 127;
  const int t = threadIdx.x;

  // Stage X[b, :, h, :] transposed to [w][c], cast to bf16.
  const float* xbase = x + (long)b * Cc * Hc * Wc + (long)h * Wc;
  for (int idx = t; idx < Cc * Wc; idx += 256) {
    int c = idx >> 7, w = idx & 127;
    xlds[w * LDW + c] = (bf16)xbase[(long)c * Hc * Wc + w];
  }
  for (int idx = t; idx < Rc * Cc; idx += 256) {
    int r = idx >> 8, c = idx & 255;
    wq[r * LDW + c] = (bf16)Wq[idx];
    wk[r * LDW + c] = (bf16)Wk[idx];
  }
  for (int idx = t; idx < Cc * Cc; idx += 256) {
    int r = idx >> 8, c = idx & 255;
    wv[r * LDW + c] = (bf16)Wv[idx];
  }
  __syncthreads();

  const int lane = t & 31, wid = t >> 5;
  const int wbase = wid * 16;                 // this wave's 16-pixel M tile
  const int n = lane & 15, hi = lane >> 4;
  const long pixbase = (long)(b * Hc + h) * Wc + wbase;

  // ---- q and k: M=16 pixels, N=32 (2 tiles), K=256 (8 chunks) ----
  v8f qa0 = {}, qa1 = {}, ka0 = {}, ka1 = {};
  for (int kc = 0; kc < 8; ++kc) {
    v16bf A = frag_a(xlds + wbase * LDW, LDW, kc * 32, lane);
    qa0 = wmma_bf16(A, frag_b(wq, LDW, kc * 32, lane), qa0);
    qa1 = wmma_bf16(A, frag_b(wq + 16 * LDW, LDW, kc * 32, lane), qa1);
    ka0 = wmma_bf16(A, frag_b(wk, LDW, kc * 32, lane), ka0);
    ka1 = wmma_bf16(A, frag_b(wk + 16 * LDW, LDW, kc * 32, lane), ka1);
  }
  const float bq0 = bq[n], bq1 = bq[16 + n], bk0 = bk[n], bk1 = bk[16 + n];
#pragma unroll
  for (int r = 0; r < 8; ++r) {
    const int m = r + hi * 8;
    const long pq = (pixbase + m) * Rc;
    qws[pq + n] = (bf16)(qa0[r] + bq0);
    qws[pq + 16 + n] = (bf16)(qa1[r] + bq1);
    kws[pq + n] = (bf16)(ka0[r] + bk0);
    kws[pq + 16 + n] = (bf16)(ka1[r] + bk1);
  }

  // ---- v: M=16 pixels, N=256 (16 tiles), K=256 (8 chunks) ----
  v8f va[16] = {};
  for (int kc = 0; kc < 8; ++kc) {
    v16bf A = frag_a(xlds + wbase * LDW, LDW, kc * 32, lane);
#pragma unroll
    for (int nt = 0; nt < 16; ++nt)
      va[nt] = wmma_bf16(A, frag_b(wv + nt * 16 * LDW, LDW, kc * 32, lane), va[nt]);
  }
#pragma unroll
  for (int nt = 0; nt < 16; ++nt) {
    const float bb = bv[nt * 16 + n];
#pragma unroll
    for (int r = 0; r < 8; ++r) {
      const int m = r + hi * 8;
      vws[(pixbase + m) * Cc + nt * 16 + n] = (bf16)(va[nt][r] + bb);
    }
  }
}

// ---------------------------------------------------------------------------
// Kernels 2/3 shared body: one line of 128 tokens, scores -> softmax -> @V.
// axis==0: row attention, block = (b,h), tokens indexed by w, writes rowws.
// axis==1: col attention, block = (b,w), tokens indexed by h, fuses
//          combined = out_col + rowws (RMW) and attn_map channel-mean.
// ---------------------------------------------------------------------------
template <int AXIS>
__global__ void axial_attn_kernel(const bf16* __restrict__ qws,
                                  const bf16* __restrict__ kws,
                                  const bf16* __restrict__ vws,
                                  float* __restrict__ rowws,
                                  float* __restrict__ attn) {
  extern __shared__ char smem[];
  bf16* qld = (bf16*)smem;                        // [128][LDQ]
  bf16* kld = qld + 128 * LDQ;                    // [128][LDQ]
  bf16* vT = kld + 128 * LDQ;                     // [256][LDV]  (N-major for frag_b)
  float* sld = (float*)(vT + 256 * LDV);          // [8 waves][16][LDS_SC]
  bf16* pld = (bf16*)(sld + 8 * 16 * LDS_SC);     // [8 waves][16][LDV]
  float* rinv = (float*)(pld + 8 * 16 * LDV);     // [8 waves][16]

  const int bid = blockIdx.x;
  const int b = bid >> 7, fixed = bid & 127;  // fixed = h (row) or w (col)
  const int t = threadIdx.x;

  // token i -> global pixel index
  auto pix = [&](int i) -> long {
    if (AXIS == 0) return (long)(b * Hc + fixed) * Wc + i;  // vary w
    return (long)(b * Hc + i) * Wc + fixed;                  // vary h
  };

  for (int idx = t; idx < 128 * Rc; idx += 256) {
    int i = idx >> 5, r = idx & 31;
    const long p = pix(i) * Rc;
    qld[i * LDQ + r] = qws[p + r];
    kld[i * LDQ + r] = kws[p + r];
  }
  for (int idx = t; idx < 128 * Cc; idx += 256) {
    int u = idx >> 8, c = idx & 255;
    vT[c * LDV + u] = vws[pix(u) * Cc + c];
  }
  __syncthreads();

  const int lane = t & 31, wid = t >> 5;
  const int n = lane & 15, hi = lane >> 4;
  const int mb = wid * 16;  // this wave's 16 query tokens
  float* sw = sld + wid * 16 * LDS_SC;
  bf16* pw = pld + wid * 16 * LDV;

  // ---- scores: [16 x 128], K = R = 32 (single chunk) ----
  {
    v16bf A = frag_a(qld + mb * LDQ, LDQ, 0, lane);
#pragma unroll
    for (int nt = 0; nt < 8; ++nt) {
      v8f s = {};
      s = wmma_bf16(A, frag_b(kld + nt * 16 * LDQ, LDQ, 0, lane), s);
#pragma unroll
      for (int r = 0; r < 8; ++r) sw[(r + hi * 8) * LDS_SC + nt * 16 + n] = s[r];
    }
  }
  __syncthreads();

  // ---- softmax along 128: 2 lanes per row, 1/sum deferred to epilogue ----
  {
    const int row = lane >> 1, half = lane & 1;
    const float* sp = sw + row * LDS_SC + half * 64;
    bf16* pp = pw + row * LDV + half * 64;
    float mx = -3.4e38f;
    for (int i = 0; i < 64; ++i) mx = fmaxf(mx, sp[i]);
    mx = fmaxf(mx, __shfl_xor(mx, 1));
    float sum = 0.f;
    for (int i = 0; i < 64; ++i) {
      float e = __expf(sp[i] - mx);
      sum += e;
      pp[i] = (bf16)e;
    }
    sum += __shfl_xor(sum, 1);
    if (half == 0) rinv[wid * 16 + row] = 1.f / sum;
  }
  __syncthreads();

  // ---- out = P @ V: [16 x 256], K = 128 (4 chunks) ----
  v8f acc[16] = {};
  for (int kc = 0; kc < 4; ++kc) {
    v16bf A = frag_a(pw, LDV, kc * 32, lane);
#pragma unroll
    for (int nt = 0; nt < 16; ++nt)
      acc[nt] = wmma_bf16(A, frag_b(vT + nt * 16 * LDV, LDV, kc * 32, lane), acc[nt]);
  }

  float rv[8];
#pragma unroll
  for (int r = 0; r < 8; ++r) rv[r] = rinv[wid * 16 + r + hi * 8];

  if (AXIS == 0) {
    // row pass: store normalized out_row (f32) to workspace, channels-last.
#pragma unroll
    for (int nt = 0; nt < 16; ++nt)
#pragma unroll
      for (int r = 0; r < 8; ++r) {
        const int m = r + hi * 8;
        rowws[pix(mb + m) * Cc + nt * 16 + n] = acc[nt][r] * rv[r];
      }
  } else {
    // col pass: combined = out_col + out_row (RMW) + attn_map channel-mean.
    float asum[8] = {};
#pragma unroll
    for (int nt = 0; nt < 16; ++nt)
#pragma unroll
      for (int r = 0; r < 8; ++r) {
        const int m = r + hi * 8;
        const long gp = pix(mb + m) * Cc + nt * 16 + n;
        const float comb = acc[nt][r] * rv[r] + rowws[gp];
        rowws[gp] = comb;
        asum[r] += comb;
      }
#pragma unroll
    for (int r = 0; r < 8; ++r) {
      float s = asum[r];
      s += __shfl_xor(s, 1, 16);
      s += __shfl_xor(s, 2, 16);
      s += __shfl_xor(s, 4, 16);
      s += __shfl_xor(s, 8, 16);
      if (n == 0) {
        const int hh = mb + r + hi * 8;
        attn[(long)b * Hc * Wc + (long)hh * Wc + fixed] = s * (0.5f / (float)Cc);
      }
    }
  }
}

// ---------------------------------------------------------------------------
// Kernel 4: out[b,c,h,w] = x + gamma*0.5*comb[b,(h,w),c] via 32x32 LDS transpose.
// ---------------------------------------------------------------------------
__global__ void finalize_kernel(const float* __restrict__ x,
                                const float* __restrict__ comb,
                                const float* __restrict__ gamma,
                                float* __restrict__ out) {
  __shared__ float tile[32][33];
  const int b = blockIdx.z;
  const int p0 = blockIdx.x * 32;  // pixel tile (h*W + w)
  const int c0 = blockIdx.y * 32;  // channel tile
  const int tx = threadIdx.x, ty = threadIdx.y;  // (32, 8)
  const float* cb = comb + (long)b * Hc * Wc * Cc;
#pragma unroll
  for (int i = 0; i < 32; i += 8)
    tile[ty + i][tx] = cb[(long)(p0 + ty + i) * Cc + c0 + tx];
  __syncthreads();
  const float g = gamma[0] * 0.5f;
  const float* xb = x + (long)b * Cc * Hc * Wc;
  float* ob = out + (long)b * Cc * Hc * Wc;
#pragma unroll
  for (int i = 0; i < 32; i += 8) {
    const long o = (long)(c0 + ty + i) * Hc * Wc + p0 + tx;
    ob[o] = xb[o] + g * tile[tx][ty + i];
  }
}

// ---------------------------------------------------------------------------
extern "C" void kernel_launch(void* const* d_in, const int* in_sizes, int n_in,
                              void* d_out, int out_size, void* d_ws, size_t ws_size,
                              hipStream_t stream) {
  const float* x = (const float*)d_in[0];
  const float* Wq = (const float*)d_in[1];
  const float* bq = (const float*)d_in[2];
  const float* Wk = (const float*)d_in[3];
  const float* bk = (const float*)d_in[4];
  const float* Wv = (const float*)d_in[5];
  const float* bv = (const float*)d_in[6];
  const float* gamma = (const float*)d_in[7];

  float* out = (float*)d_out;
  float* attn = out + (size_t)Bc * Cc * Hc * Wc;  // attn_map tail of d_out

  // Workspace carve-up:
  //   qws  bf16 [B,H,W,R]   8 MiB
  //   kws  bf16 [B,H,W,R]   8 MiB
  //   vws  bf16 [B,H,W,C]  64 MiB
  //   rowws f32 [B,H,W,C] 128 MiB  (out_row, then combined)
  char* ws = (char*)d_ws;
  const size_t nqk = (size_t)Bc * Hc * Wc * Rc;
  const size_t nv = (size_t)Bc * Hc * Wc * Cc;
  bf16* qws = (bf16*)ws;
  bf16* kws = (bf16*)(ws + nqk * 2);
  bf16* vws = (bf16*)(ws + 2 * nqk * 2);
  float* rowws = (float*)(ws + 2 * nqk * 2 + nv * 2);

  const size_t lds1 =
      ((size_t)Wc * LDW + 2 * Rc * LDW + (size_t)Cc * LDW) * sizeof(bf16);
  const size_t lds2 = (size_t)2 * 128 * LDQ * 2 + (size_t)256 * LDV * 2 +
                      (size_t)8 * 16 * LDS_SC * 4 + (size_t)8 * 16 * LDV * 2 +
                      (size_t)8 * 16 * 4;

  qkv_kernel<<<Bc * Hc, 256, lds1, stream>>>(x, Wq, bq, Wk, bk, Wv, bv, qws, kws, vws);
  axial_attn_kernel<0><<<Bc * Hc, 256, lds2, stream>>>(qws, kws, vws, rowws, attn);
  axial_attn_kernel<1><<<Bc * Wc, 256, lds2, stream>>>(qws, kws, vws, rowws, attn);
  finalize_kernel<<<dim3((Hc * Wc) / 32, Cc / 32, Bc), dim3(32, 8), 0, stream>>>(
      x, rowws, gamma, out);
}
